// _TinySSM_74552042324816
// MI455X (gfx1250) — matrix operand order repaired and checked by reading, not compiled, not run on hardware
//
#include <hip/hip_runtime.h>
#include <hip/hip_bf16.h>

// ---------------------------------------------------------------------------
// TinySSM forward for MI455X (gfx1250, wave32, WMMA, async global->LDS).
//  dims: B=4, L=4096, D_INNER=D_MODEL=2048, D_STATE=16, M = B*L = 16384
// ---------------------------------------------------------------------------

#define BATCH    4
#define SEQLEN   4096
#define DI       2048
#define DS       16
#define MROWS    (BATCH * SEQLEN)   // 16384

typedef __attribute__((ext_vector_type(16))) __bf16 v16bf;
typedef __attribute__((ext_vector_type(8)))  float  v8f;

union FragU { uint4 q[2]; unsigned int u[8]; v16bf v; };

// round-to-nearest-even f32 -> bf16 (raw bits)
static __device__ __forceinline__ unsigned short f32_to_bf16(float f) {
    unsigned int u = __float_as_uint(f);
    u += 0x7FFFu + ((u >> 16) & 1u);
    return (unsigned short)(u >> 16);
}
static __device__ __forceinline__ float bf16_to_f32(unsigned short h) {
    return __uint_as_float(((unsigned int)h) << 16);
}

// CDNA5 async global->LDS copy of 16 bytes (tracked by ASYNCcnt, in-order done).
static __device__ __forceinline__ void async_copy_b128(unsigned int lds_addr,
                                                       unsigned long long gaddr) {
    asm volatile("global_load_async_to_lds_b128 %0, %1, off"
                 :: "v"(lds_addr), "v"(gaddr) : "memory");
}
static __device__ __forceinline__ void wait_async0() {
    asm volatile("s_wait_asynccnt 0x0" ::: "memory");
}
// allow the newest stage (4 copies) to remain in flight
static __device__ __forceinline__ void wait_async4() {
    asm volatile("s_wait_asynccnt 0x4" ::: "memory");
}

// ---------------------------------------------------------------------------
// f32 -> bf16 conversion (grid-stride)
// ---------------------------------------------------------------------------
__global__ __launch_bounds__(256) void cvt_bf16_kernel(const float* __restrict__ in,
                                                       unsigned short* __restrict__ out,
                                                       int n) {
    int i = blockIdx.x * 256 + threadIdx.x;
    int stride = gridDim.x * 256;
    for (; i < n; i += stride) out[i] = f32_to_bf16(in[i]);
}

// ---------------------------------------------------------------------------
// Big bf16 WMMA GEMM:  out[m,n] = sum_k A[m,k] * W[n,k] + bias[n]
//   Block: 256 threads = 8 waves; block tile 128(M) x 128(N); K-chunk 32.
//   Wave grid 4(M) x 2(N); each wave computes 32x64 = 2x4 WMMA tiles.
//   Triple-buffered LDS, depth-2 async global->LDS pipeline with partial
//   s_wait_asynccnt so copies overlap the full WMMA phase.
//   mode 0: write f32; mode 1: softplus -> bf16.
// ---------------------------------------------------------------------------
#define GTM 128
#define GTN 128
#define GTK 32
#define LSTR 40   // ushorts per LDS row (32 + 8 pad)

__global__ __launch_bounds__(256) void gemm_bf16_big_kernel(
    const unsigned short* __restrict__ A,
    const unsigned short* __restrict__ W,
    const float* __restrict__ bias,
    float* __restrict__ outF,
    unsigned short* __restrict__ outH,
    int N, int K, int mode)
{
    __shared__ unsigned short sA[3][GTM * LSTR];   // 3 x 10240 B
    __shared__ unsigned short sB[3][GTN * LSTR];   // 3 x 10240 B

    const int tid  = threadIdx.x;
    const int wave = tid >> 5;
    const int lane = tid & 31;
    const int wm   = wave & 3;     // 0..3  -> M sub-tile (32 rows)
    const int wn   = wave >> 2;    // 0..1  -> N sub-tile (64 cols)

    const int m0 = blockIdx.y * GTM;
    const int n0 = blockIdx.x * GTN;

    // staging map: id = tid*2 + j in [0,512): row = id>>2 (0..127), seg = id&3
    const int sRow0 = (tid * 2 + 0) >> 2, sCol0 = ((tid * 2 + 0) & 3) * 8;
    const int sRow1 = (tid * 2 + 1) >> 2, sCol1 = ((tid * 2 + 1) & 3) * 8;

    // fragment LDS K-offsets (ISA 7.12.2 wave32 16-bit layouts):
    //  A (16x32): lane<16 -> VGPR0-3 = K0..7, VGPR4-7 = K16..23
    //             lane>=16 -> K8..15 and K24..31
    //  B (32x16): same K order as A: lane<16 -> K0..7, K16..23 ; lane>=16 -> K8..15, K24..31
    const int aK0 = (lane < 16) ? 0 : 8;
    const int aK1 = (lane < 16) ? 16 : 24;
    const int bK0 = (lane < 16) ? 0 : 8;

    v8f acc[2][4];
#pragma unroll
    for (int mi = 0; mi < 2; ++mi)
#pragma unroll
        for (int ni = 0; ni < 4; ++ni) acc[mi][ni] = (v8f){};

    auto stage = [&](int buf, int kb) {
        // A tile: 128 x 32 bf16  (2 b128 per thread)
        async_copy_b128((unsigned int)(uintptr_t)&sA[buf][sRow0 * LSTR + sCol0],
            (unsigned long long)(uintptr_t)(A + (size_t)(m0 + sRow0) * K + kb + sCol0));
        async_copy_b128((unsigned int)(uintptr_t)&sA[buf][sRow1 * LSTR + sCol1],
            (unsigned long long)(uintptr_t)(A + (size_t)(m0 + sRow1) * K + kb + sCol1));
        // B tile: 128 x 32 bf16 (rows of W; 2 b128 per thread)
        async_copy_b128((unsigned int)(uintptr_t)&sB[buf][sRow0 * LSTR + sCol0],
            (unsigned long long)(uintptr_t)(W + (size_t)(n0 + sRow0) * K + kb + sCol0));
        async_copy_b128((unsigned int)(uintptr_t)&sB[buf][sRow1 * LSTR + sCol1],
            (unsigned long long)(uintptr_t)(W + (size_t)(n0 + sRow1) * K + kb + sCol1));
    };

    const int nChunks = K / GTK;

    // depth-2 prologue
    stage(0, 0);
    if (nChunks > 1) stage(1, GTK);

    for (int kc = 0; kc < nChunks; ++kc) {
        const int cur = kc % 3;

        // async dones retire in order: <=4 outstanding means everything except
        // the newest stage (chunk kc+1) has landed -> chunk kc is ready.
        if (kc + 1 < nChunks) wait_async4();
        else                  wait_async0();
        __syncthreads();   // all waves: chunk kc visible, old reads retired

        const unsigned short* pa = sA[cur];
        const unsigned short* pb = sB[cur];

        FragU fa[2], fb[4];
#pragma unroll
        for (int mi = 0; mi < 2; ++mi) {
            const int r = (wm * 32 + mi * 16 + (lane & 15)) * LSTR;
            fa[mi].q[0] = *(const uint4*)&pa[r + aK0];
            fa[mi].q[1] = *(const uint4*)&pa[r + aK1];
        }
#pragma unroll
        for (int ni = 0; ni < 4; ++ni) {
            const int c = (wn * 64 + ni * 16 + (lane & 15)) * LSTR;
            fb[ni].q[0] = *(const uint4*)&pb[c + bK0];
            fb[ni].q[1] = *(const uint4*)&pb[c + bK0 + 16];
        }

        // issue copies for chunk kc+2 into the buffer freed at iteration kc-1
        if (kc + 2 < nChunks) stage((kc + 2) % 3, (kc + 2) * GTK);

#pragma unroll
        for (int mi = 0; mi < 2; ++mi)
#pragma unroll
            for (int ni = 0; ni < 4; ++ni)
                acc[mi][ni] = __builtin_amdgcn_wmma_f32_16x16x32_bf16(
                    false, fa[mi].v, false, fb[ni].v,
                    (short)0, acc[mi][ni], false, false);
    }

    // epilogue: C/D layout — lanes 0-15: M=r, lanes 16-31: M=8+r; N=lane&15
#pragma unroll
    for (int mi = 0; mi < 2; ++mi) {
#pragma unroll
        for (int ni = 0; ni < 4; ++ni) {
            const int col   = n0 + wn * 64 + ni * 16 + (lane & 15);
            const int mBase = m0 + wm * 32 + mi * 16 + ((lane >= 16) ? 8 : 0);
            const float bv  = bias[col];
#pragma unroll
            for (int r = 0; r < 8; ++r) {
                const size_t idx = (size_t)(mBase + r) * N + col;
                float v = acc[mi][ni][r] + bv;
                if (mode == 1) {
                    v = fmaxf(v, 0.f) + log1pf(__expf(-fabsf(v)));
                    outH[idx] = f32_to_bf16(v);
                } else {
                    outF[idx] = v;
                }
            }
        }
    }
}

// ---------------------------------------------------------------------------
// Small bf16 WMMA GEMM for N=32 (bc projection). Block tile 64x32.
// ---------------------------------------------------------------------------
#define TM 64
#define TN 32
#define TK 32

__global__ __launch_bounds__(256) void gemm_bf16_small_kernel(
    const unsigned short* __restrict__ A,
    const unsigned short* __restrict__ W,
    const float* __restrict__ bias,
    float* __restrict__ outF,
    int N, int K)
{
    __shared__ unsigned short sA[TM * LSTR];
    __shared__ unsigned short sB[TN * LSTR];

    const int tid  = threadIdx.x;
    const int wave = tid >> 5;
    const int lane = tid & 31;
    const int wm   = wave & 3;
    const int wn   = wave >> 2;

    const int m0 = blockIdx.y * TM;
    const int n0 = blockIdx.x * TN;

    const int aK0 = (lane < 16) ? 0 : 8;
    const int aK1 = (lane < 16) ? 16 : 24;
    const int bK0 = (lane < 16) ? 0 : 8;
    const int mA  = ((lane & 15) + wm * 16) * LSTR;
    const int nB  = ((lane & 15) + wn * 16) * LSTR;

    const int rA = tid >> 2;            // 0..63
    const int cA = (tid & 3) * 8;

    v8f acc = {};

    for (int kb = 0; kb < K; kb += TK) {
        async_copy_b128((unsigned int)(uintptr_t)&sA[rA * LSTR + cA],
            (unsigned long long)(uintptr_t)(A + (size_t)(m0 + rA) * K + kb + cA));
        if (tid < 128) {
            const int rB = rA & 31;
            async_copy_b128((unsigned int)(uintptr_t)&sB[rB * LSTR + cA],
                (unsigned long long)(uintptr_t)(W + (size_t)(n0 + rB) * K + kb + cA));
        }
        wait_async0();
        __syncthreads();

        FragU fa, fb;
        fa.q[0] = *(const uint4*)&sA[mA + aK0];
        fa.q[1] = *(const uint4*)&sA[mA + aK1];
        fb.q[0] = *(const uint4*)&sB[nB + bK0];
        fb.q[1] = *(const uint4*)&sB[nB + bK0 + 16];

        acc = __builtin_amdgcn_wmma_f32_16x16x32_bf16(
            false, fa.v, false, fb.v, (short)0, acc, false, false);

        __syncthreads();
    }

    const int col   = n0 + wn * 16 + (lane & 15);
    const int mBase = m0 + wm * 16 + ((lane >= 16) ? 8 : 0);
    const float bv  = bias[col];
#pragma unroll
    for (int r = 0; r < 8; ++r)
        outF[(size_t)(mBase + r) * N + col] = acc[r] + bv;
}

// ---------------------------------------------------------------------------
// Selective scan: one thread per (b,d) channel, h[16] in registers.
// ---------------------------------------------------------------------------
__global__ __launch_bounds__(256) void scan_kernel(
    const float* __restrict__ x,
    const float* __restrict__ A_log,
    const unsigned short* __restrict__ dtb,
    const float* __restrict__ bc,
    unsigned short* __restrict__ ybf)
{
    __shared__ float sbc[2 * DS];

    const int tid = threadIdx.x;
    const int d   = blockIdx.x * 256 + tid;
    const int b   = blockIdx.y;

    float a[DS], h[DS];
#pragma unroll
    for (int n = 0; n < DS; ++n) {
        a[n] = -__expf(A_log[(size_t)d * DS + n]);
        h[n] = 0.f;
    }

    for (int l = 0; l < SEQLEN; ++l) {
        const size_t row = (size_t)b * SEQLEN + l;
        if (tid < 2 * DS) sbc[tid] = bc[row * (2 * DS) + tid];
        __syncthreads();

        const float xv  = x[row * DI + d];
        const float dtv = bf16_to_f32(dtb[row * DI + d]);
        const float dtx = dtv * xv;

        float y = 0.f;
#pragma unroll
        for (int n = 0; n < DS; ++n) {
            const float dA = __expf(dtv * a[n]);
            h[n] = dA * h[n] + dtx * sbc[n];
            y += h[n] * sbc[DS + n];
        }
        ybf[row * DI + d] = f32_to_bf16(y);
        __syncthreads();
    }
}

// ---------------------------------------------------------------------------
// Host-side launcher
// ---------------------------------------------------------------------------
extern "C" void kernel_launch(void* const* d_in, const int* in_sizes, int n_in,
                              void* d_out, int out_size, void* d_ws, size_t ws_size,
                              hipStream_t stream) {
    (void)in_sizes; (void)n_in; (void)out_size; (void)ws_size;

    const float* x         = (const float*)d_in[0];
    const float* A_log     = (const float*)d_in[1];
    const float* x_proj_w  = (const float*)d_in[2];
    const float* x_proj_b  = (const float*)d_in[3];
    const float* dt_proj_w = (const float*)d_in[4];
    const float* dt_proj_b = (const float*)d_in[5];
    const float* out_proj_w= (const float*)d_in[6];
    const float* out_proj_b= (const float*)d_in[7];
    float* out = (float*)d_out;

    char* ws = (char*)d_ws;
    unsigned short* xb   = (unsigned short*)ws; ws += (size_t)MROWS * DI * 2;
    unsigned short* wdt  = (unsigned short*)ws; ws += (size_t)DI * DI * 2;
    unsigned short* wxp  = (unsigned short*)ws; ws += (size_t)(2 * DS) * DI * 2;
    unsigned short* wout = (unsigned short*)ws; ws += (size_t)DI * DI * 2;
    unsigned short* dtb  = (unsigned short*)ws; ws += (size_t)MROWS * DI * 2;
    unsigned short* ybf  = (unsigned short*)ws; ws += (size_t)MROWS * DI * 2;
    float* bc            = (float*)ws;

    // 1) conversions to bf16
    {
        int n = MROWS * DI;
        cvt_bf16_kernel<<<(n + 255) / 256, 256, 0, stream>>>(x, xb, n);
        n = DI * DI;
        cvt_bf16_kernel<<<(n + 255) / 256, 256, 0, stream>>>(dt_proj_w, wdt, n);
        cvt_bf16_kernel<<<(n + 255) / 256, 256, 0, stream>>>(out_proj_w, wout, n);
        n = 2 * DS * DI;
        cvt_bf16_kernel<<<(n + 255) / 256, 256, 0, stream>>>(x_proj_w, wxp, n);
    }
    // 2) dt = softplus(x @ dtW^T + dt_b) -> bf16
    {
        dim3 grid(DI / GTN, MROWS / GTM);
        gemm_bf16_big_kernel<<<grid, 256, 0, stream>>>(xb, wdt, dt_proj_b,
                                                       nullptr, dtb, DI, DI, 1);
    }
    // 3) bc = x @ xpW^T + xp_b -> f32  (N = 32)
    {
        dim3 grid((2 * DS) / TN, MROWS / TM);
        gemm_bf16_small_kernel<<<grid, 256, 0, stream>>>(xb, wxp, x_proj_b,
                                                         bc, 2 * DS, DI);
    }
    // 4) selective scan -> y (bf16)
    {
        dim3 grid(DI / 256, BATCH);
        scan_kernel<<<grid, 256, 0, stream>>>(x, A_log, dtb, bc, ybf);
    }
    // 5) out = y @ outW^T + out_b -> f32
    {
        dim3 grid(DI / GTN, MROWS / GTM);
        gemm_bf16_big_kernel<<<grid, 256, 0, stream>>>(ybf, wout, out_proj_b,
                                                       out, nullptr, DI, DI, 0);
    }
}
